// Atom_embedding_MP_62672162783255
// MI455X (gfx1250) — compile-verified, matrix operand order checked by reading, not run on hardware
//
#include <hip/hip_runtime.h>

typedef __attribute__((ext_vector_type(16))) _Float16 v16h;
typedef __attribute__((ext_vector_type(8)))  float    v8f;
typedef __attribute__((ext_vector_type(2)))  float    v2f;

namespace {
constexpr int   Kn    = 16;    // neighbors
constexpr int   Dd    = 6;     // embedding dim
constexpr int   Ff    = 13;    // feature dim 2D+1
constexpr int   Ll    = 3;     // layers
constexpr int   PPW   = 16;    // points per wave (weight-fragment amortization)
constexpr float SLOPE = 0.2f;
constexpr float EPS   = 1e-5f;
}

// DPP16 xor-style butterfly add step (no LDS traffic, single v_add_f32_dpp).
template <int CTRL>
__device__ __forceinline__ float dpp_add(float x) {
    int xi = __builtin_bit_cast(int, x);
    int yi = __builtin_amdgcn_update_dpp(xi, xi, CTRL, 0xF, 0xF, true);
    return x + __builtin_bit_cast(float, yi);
}

// Sum across the 16-lane row this lane belongs to. Steps: quad xor1, quad xor2,
// then (values now quad-uniform) half-mirror and row-mirror complete the sum.
__device__ __forceinline__ float rowsum16(float x) {
    x = dpp_add<0xB1>(x);   // quad_perm(1,0,3,2)  : xor 1
    x = dpp_add<0x4E>(x);   // quad_perm(2,3,0,1)  : xor 2
    x = dpp_add<0x141>(x);  // ROW_HALF_MIRROR     : xor 4 (quad-uniform)
    x = dpp_add<0x140>(x);  // ROW_MIRROR          : xor 8 (8-group-uniform)
    return x;
}

__device__ __forceinline__ float leaky(float x) {
    return fmaxf(x, x * SLOPE);      // valid since SLOPE > 0
}

// One wave32 per PPW points. Weight + bias fragments for all 3 layers are built
// once per wave and reused across points. Transposed matmuls make the C-layout
// output of WMMA #1 feed directly (lane-for-lane) as the B-fragment of WMMA #2;
// biases ride in the WMMA accumulator operand.
__global__ __launch_bounds__(256) void atom_mp_kernel(
    const float* __restrict__ dist,  // [B,N,K,1]
    const float* __restrict__ at,    // [B,N,K,D]
    const float* __restrict__ w1,    // [L,F,F]
    const float* __restrict__ b1,    // [L,F]
    const float* __restrict__ w2,    // [L,F,D]
    const float* __restrict__ b2,    // [L,D]
    const float* __restrict__ gnw,   // [L,D]
    const float* __restrict__ gnb,   // [L,D]
    float* __restrict__ out,         // [B,N,D]
    int npts)
{
    const int wave = blockIdx.x * (blockDim.x >> 5) + (threadIdx.x >> 5);
    const int base = wave * PPW;
    if (base >= npts) return;                    // wave-uniform: EXEC stays all-ones
    const int pend = (base + PPW < npts) ? base + PPW : npts;

    const int  lane = threadIdx.x & 31;
    const int  m    = lane & 15;                 // neighbor row / weight row-in-group
    const bool lo   = lane < 16;
    const int  kbase = lo ? 0 : 8;               // K-offset of this lane group in frags

    // clamped column indices so weight loads are unconditional (no exec branching)
    const int mF = (m < Ff) ? m : 0;             // valid column for w1^T rows
    const int mD = (m < Dd) ? m : 0;             // valid column for w2^T rows

    // ---- Build all per-layer weight + bias fragments ONCE per wave ----
    v16h  a1f[Ll], a2f[Ll];
    v8f   cb1f[Ll], cb2f[Ll];
    float gwv[Ll][Dd], gbv[Ll][Dd];
#pragma unroll
    for (int layer = 0; layer < Ll; ++layer) {
        const float* W1 = w1 + layer * Ff * Ff;
        const float* W2 = w2 + layer * Ff * Dd;
        const float* B1 = b1 + layer * Ff;

        v16h a1;
#pragma unroll
        for (int i = 0; i < 16; ++i) a1[i] = (_Float16)0.0f;
#pragma unroll
        for (int r = 0; r < 8; ++r) {
            const int k  = kbase + r;
            const int kc = (k < Ff) ? k : 0;
            float v = W1[kc * Ff + mF];          // unconditional, clamped address
            v = (k < Ff && m < Ff) ? v : 0.0f;   // v_cndmask, no branch
            a1[r] = (_Float16)v;
        }
        a1f[layer] = a1;

        v16h a2;
#pragma unroll
        for (int i = 0; i < 16; ++i) a2[i] = (_Float16)0.0f;
#pragma unroll
        for (int r = 0; r < 8; ++r) {
            const int k  = kbase + r;
            const int kc = (k < Ff) ? k : 0;
            float v = W2[kc * Dd + mD];
            v = (k < Ff && m < Dd) ? v : 0.0f;
            a2[r] = (_Float16)v;
        }
        a2f[layer] = a2;

        // C operand for WMMA #1: C[f][n] = b1[f] broadcast over columns.
        v8f cb1;
#pragma unroll
        for (int r = 0; r < 8; ++r) {
            const int f  = kbase + r;
            const int fc = (f < Ff) ? f : 0;
            float v = B1[fc];
            cb1[r] = (f < Ff) ? v : 0.0f;
        }
        cb1f[layer] = cb1;

        // C operand for WMMA #2: C[d][n] = b2[d]; after the 16-column rowsum
        // this contributes exactly Kn*b2[d].
        v8f cb2;
#pragma unroll
        for (int r = 0; r < 8; ++r) {
            float v = (r < Dd) ? b2[layer * Dd + r] : 0.0f;   // uniform load
            cb2[r] = lo ? v : 0.0f;              // rows 8..15 (hi group) are zero
        }
        cb2f[layer] = cb2;

#pragma unroll
        for (int d = 0; d < Dd; ++d) {           // uniform -> scalarized by compiler
            gwv[layer][d] = gnw[layer * Dd + d];
            gbv[layer][d] = gnb[layer * Dd + d];
        }
    }

#pragma unroll 1
    for (int p = base; p < pend; ++p) {
        const float* __restrict__ atp = at   + (size_t)p * (Kn * Dd);
        const float* __restrict__ dp  = dist + (size_t)p * Kn;

        // ---- Branchless build of feat^T B-fragment (32x16 f16) ----
        // lo lanes need atp[m*6+0..1] -> elems 6,7 ; hi lanes need
        // atp[m*6+2..5], dp[m] -> elems 0..4. Shared clamped addresses; the
        // slots that are garbage for lo lanes (0..5) are overwritten by the
        // per-layer pe refresh below.
        const float* ab = atp + m * Dd + (lo ? 0 : 2);
        const float v0 = __builtin_nontemporal_load(ab + 0);
        const float v1 = __builtin_nontemporal_load(ab + 1);
        const float v2 = __builtin_nontemporal_load(ab + 2);
        const float v3 = __builtin_nontemporal_load(ab + 3);
        const float dv = __builtin_nontemporal_load(dp + (lo ? 0 : m));

        v16h featF;
        featF[0] = (_Float16)v0;
        featF[1] = (_Float16)v1;
        featF[2] = (_Float16)v2;
        featF[3] = (_Float16)v3;
        featF[4] = (_Float16)dv;
        featF[5] = (_Float16)0.0f;
        featF[6] = lo ? (_Float16)v0 : (_Float16)0.0f;
        featF[7] = lo ? (_Float16)v1 : (_Float16)0.0f;
#pragma unroll
        for (int i = 8; i < 16; ++i) featF[i] = (_Float16)0.0f;

        float pe[Dd];
#pragma unroll
        for (int d = 0; d < Dd; ++d) pe[d] = 1.0f;

#pragma unroll
        for (int layer = 0; layer < Ll; ++layer) {
            // refresh point-embedding part (K=0..5, lanes 0-15) via selects
#pragma unroll
            for (int d = 0; d < Dd; ++d)
                featF[d] = lo ? (_Float16)pe[d] : featF[d];

            // h^T = w1^T @ feat^T + b1 : lane = neighbor col, vgpr r = feature
            v8f c1 = __builtin_amdgcn_wmma_f32_16x16x32_f16(
                false, a1f[layer], false, featF, (short)0, cb1f[layer], false, false);

            // leaky-relu; C-layout of h^T == B-fragment of h^T (32x16)
            v16h hF;
#pragma unroll
            for (int i = 8; i < 16; ++i) hF[i] = (_Float16)0.0f;
#pragma unroll
            for (int r = 0; r < 8; ++r)
                hF[r] = (_Float16)leaky(c1[r]);

            // msg^T = w2^T @ h^T + b2 : lane n holds neighbor n's message
            v8f c2 = __builtin_amdgcn_wmma_f32_16x16x32_f16(
                false, a2f[layer], false, hF, (short)0, cb2f[layer], false, false);

            // neighbor sum via DPP butterfly within each 16-lane row;
            // bias term sums to Kn*b2[d] automatically.
            float msg[Dd];
#pragma unroll
            for (int d = 0; d < Dd; ++d)
                msg[d] = rowsum16(c2[d]);

            // GroupNorm(2 groups of 3) + leaky-relu + residual
#pragma unroll
            for (int g = 0; g < 2; ++g) {
                const float x0 = msg[3 * g + 0], x1 = msg[3 * g + 1], x2 = msg[3 * g + 2];
                const float mu = (x0 + x1 + x2) * (1.0f / 3.0f);
                const float d0 = x0 - mu, d1 = x1 - mu, d2 = x2 - mu;
                const float var = (d0 * d0 + d1 * d1 + d2 * d2) * (1.0f / 3.0f);
                const float is  = rsqrtf(var + EPS);
#pragma unroll
                for (int j = 0; j < 3; ++j) {
                    const int d = 3 * g + j;
                    const float xn = (msg[d] - mu) * is;
                    const float y  = fmaf(xn, gwv[layer][d], gbv[layer][d]);
                    pe[d] += leaky(y);
                }
            }
        }

        if (lane == 0) {
            v2f* op = (v2f*)(out + (size_t)p * Dd);   // 8B aligned (24B stride)
            v2f s0 = {pe[0], pe[1]};
            v2f s1 = {pe[2], pe[3]};
            v2f s2 = {pe[4], pe[5]};
            __builtin_nontemporal_store(s0, op + 0);
            __builtin_nontemporal_store(s1, op + 1);
            __builtin_nontemporal_store(s2, op + 2);
        }
    }
}

extern "C" void kernel_launch(void* const* d_in, const int* in_sizes, int n_in,
                              void* d_out, int out_size, void* d_ws, size_t ws_size,
                              hipStream_t stream) {
    const float* dist = (const float*)d_in[0];
    const float* at   = (const float*)d_in[1];
    const float* w1   = (const float*)d_in[2];
    const float* b1   = (const float*)d_in[3];
    const float* w2   = (const float*)d_in[4];
    const float* b2   = (const float*)d_in[5];
    const float* gnw  = (const float*)d_in[6];
    const float* gnb  = (const float*)d_in[7];
    float* out = (float*)d_out;

    const int npts  = in_sizes[0] / Kn;          // B*N (dist has B*N*K elements)
    const int waves = (npts + PPW - 1) / PPW;
    const int wavesPerBlock = 8;                 // 256 threads = 8 wave32
    const int blocks = (waves + wavesPerBlock - 1) / wavesPerBlock;
    atom_mp_kernel<<<blocks, 256, 0, stream>>>(dist, at, w1, b1, w2, b2,
                                               gnw, gnb, out, npts);
}
